// SovereignLeviathanV2_63307817943081
// MI455X (gfx1250) — matrix operand order repaired and checked by reading, not compiled, not run on hardware
//
#include <hip/hip_runtime.h>
#include <math.h>

// ---------------------------------------------------------------------------
// Sizes (fixed by the reference)
// ---------------------------------------------------------------------------
#define D_MODEL 512
#define D_FF    2048
#define N_EXP   8
#define VOCAB   256
#define BATCH   4
#define TSEQ    2048
#define NTOK    (BATCH * TSEQ)     // 8192

#if defined(__gfx1250__)
#define USE_ASYNC_LDS 1
#else
#define USE_ASYNC_LDS 0
#endif

typedef __attribute__((ext_vector_type(16))) __bf16 v16bf;
typedef __attribute__((ext_vector_type(8)))  float  v8f;

// ---------------------------------------------------------------------------
// bf16 helpers (round-to-nearest-even-ish)
// ---------------------------------------------------------------------------
__device__ __forceinline__ unsigned short f2bf(float f) {
    unsigned int u = __builtin_bit_cast(unsigned int, f);
    u += 0x7FFFu + ((u >> 16) & 1u);
    return (unsigned short)(u >> 16);
}
__device__ __forceinline__ float bf2f(unsigned short h) {
    unsigned int u = ((unsigned int)h) << 16;
    return __builtin_bit_cast(float, u);
}

#if USE_ASYNC_LDS
// 32-byte global->LDS async copy (2 x b128). Generic LDS pointer low 32 bits
// are the wave-relative LDS byte offset (ISA 10.2 aperture mapping).
// instruction offset is added to BOTH the LDS and global addresses.
__device__ __forceinline__ void async_copy32(const void* gsrc, void* lds_dst) {
    unsigned lo = (unsigned)(size_t)lds_dst;
    unsigned long long ga = (unsigned long long)gsrc;
    asm volatile(
        "global_load_async_to_lds_b128 %0, %1, off\n\t"
        "global_load_async_to_lds_b128 %0, %1, off offset:16"
        :: "v"(lo), "v"(ga) : "memory");
}
__device__ __forceinline__ void async_wait0() {
    asm volatile("s_wait_asynccnt 0" ::: "memory");
}
#endif

// ---------------------------------------------------------------------------
// Weight pre-pass: out[e][n][k] = bf16(in[e][k][n])  (transpose + convert)
// ---------------------------------------------------------------------------
__global__ __launch_bounds__(256)
void transpose_to_bf16(const float* __restrict__ in, unsigned short* __restrict__ out,
                       int K, int N) {
    size_t i  = (size_t)blockIdx.x * 256 + threadIdx.x;   // < nmat*N*K
    size_t nk = (size_t)N * K;
    size_t e  = i / nk;
    size_t r  = i - e * nk;
    int n = (int)(r / K), k = (int)(r % K);
    out[i] = f2bf(in[e * nk + (size_t)k * N + n]);
}

// ---------------------------------------------------------------------------
// Embedding lookup -> bf16 X [NTOK, D_MODEL]
// ---------------------------------------------------------------------------
__global__ __launch_bounds__(256)
void embed_kernel(const int* __restrict__ seq, const float* __restrict__ emb,
                  unsigned short* __restrict__ xe) {
    size_t i = (size_t)blockIdx.x * blockDim.x + threadIdx.x;   // < NTOK*D_MODEL
    int m = (int)(i >> 9);
    int c = (int)(i & 511);
    int tok = seq[m];
    xe[i] = f2bf(emb[(size_t)tok * D_MODEL + c]);
}

// ---------------------------------------------------------------------------
// Toroidal scan: 2048 independent (b,c) chains over T=2048 steps.
// ---------------------------------------------------------------------------
__global__ __launch_bounds__(256)
void scan_kernel(const float* __restrict__ ANG, const float* __restrict__ AMP,
                 unsigned short* __restrict__ X1, float* __restrict__ nstate) {
    int tid = blockIdx.x * blockDim.x + threadIdx.x;   // < BATCH*D_MODEL
    int b = tid >> 9;
    int c = tid & 511;
    float state = 0.0f;
    for (int t = 0; t < TSEQ; ++t) {
        size_t idx = ((size_t)(b * TSEQ + t)) * D_MODEL + c;
        float a   = ANG[idx];
        float amp = AMP[idx];
        float si, co;
        __sincosf(a, &si, &co);
        float s = co * state - si * (1.0f - state);
        s = fminf(1.0f, fmaxf(-1.0f, s));
        X1[idx] = f2bf(amp * s);
        state = s;
    }
    nstate[(size_t)b * D_MODEL + c] = state;
}

// ---------------------------------------------------------------------------
// Top-2 softmax routing -> dense per-expert weights WE [NTOK, N_EXP]
// ---------------------------------------------------------------------------
__global__ __launch_bounds__(256)
void gate_kernel(const unsigned short* __restrict__ X1,
                 const float* __restrict__ gw, const float* __restrict__ gb,
                 float* __restrict__ WE) {
    int m = blockIdx.x * blockDim.x + threadIdx.x;     // < NTOK
    float acc[N_EXP];
#pragma unroll
    for (int e = 0; e < N_EXP; ++e) acc[e] = gb[e];
    for (int k = 0; k < D_MODEL; ++k) {
        float x = bf2f(X1[(size_t)m * D_MODEL + k]);
#pragma unroll
        for (int e = 0; e < N_EXP; ++e) acc[e] += x * gw[k * N_EXP + e];
    }
    float mx = acc[0];
#pragma unroll
    for (int e = 1; e < N_EXP; ++e) mx = fmaxf(mx, acc[e]);
#pragma unroll
    for (int e = 0; e < N_EXP; ++e) acc[e] = __expf(acc[e] - mx);
    int   i0 = 0; float v0 = acc[0];
#pragma unroll
    for (int e = 1; e < N_EXP; ++e) if (acc[e] > v0) { v0 = acc[e]; i0 = e; }
    int   i1 = -1; float v1 = -1.0f;
#pragma unroll
    for (int e = 0; e < N_EXP; ++e)
        if (e != i0 && acc[e] > v1) { v1 = acc[e]; i1 = e; }
    float inv = 1.0f / (v0 + v1);
#pragma unroll
    for (int e = 0; e < N_EXP; ++e) {
        float w = (e == i0) ? v0 * inv : ((e == i1) ? v1 * inv : 0.0f);
        WE[(size_t)m * N_EXP + e] = w;
    }
}

// ---------------------------------------------------------------------------
// Tiled WMMA GEMM: C[M,N] = A[M,K] @ B[K,N] with B supplied pre-transposed and
// pre-converted: Bt[n][k] bf16.  A is bf16 (A_BF16) or fp32 (head).
// Block: 256 threads = 8 waves; tile 128x128, BK = 32.
// Wave (wr,wc) owns a 64x32 region = 4x2 16x16 tiles.
// Double-buffered LDS; A/B tiles staged with async global->LDS DMA (ASYNCcnt)
// overlapping the WMMA pipe; fp32 A falls back to convert-on-store staging.
// ---------------------------------------------------------------------------
enum { EPI_ANG = 0, EPI_AMP, EPI_GELU, EPI_MOE, EPI_HEAD };

template <bool A_BF16, int EPI>
__global__ __launch_bounds__(256)
void gemm_wmma(const void* __restrict__ Ap, const unsigned short* __restrict__ Bt,
               const float* __restrict__ bias, const float* __restrict__ we,
               void* __restrict__ Out, int M, int N, int K) {
    __shared__ unsigned short Alds[2][128 * 32];
    __shared__ unsigned short Blds[2][128 * 32];

    const int tid  = threadIdx.x;
    const int lane = tid & 31;
    const int wv   = tid >> 5;
    const int wr   = wv >> 2;          // 0..1
    const int wc   = wv & 3;           // 0..3
    const int m0   = blockIdx.y * 128;
    const int n0   = blockIdx.x * 128;
    const int ar   = tid >> 1, ah = tid & 1;   // staging: row, 16-elem half
    const int half = lane >> 4, l16 = lane & 15;

    v8f acc[4][2];
#pragma unroll
    for (int i = 0; i < 4; ++i)
#pragma unroll
        for (int j = 0; j < 2; ++j)
            acc[i][j] = (v8f){0.f, 0.f, 0.f, 0.f, 0.f, 0.f, 0.f, 0.f};

    const int kstages = K >> 5;
    const int ldsOff  = ar * 32 + ah * 16;          // element offset in tile

    // ---------------- prologue: stage tile 0 into buffer 0 ----------------
    {
#if USE_ASYNC_LDS
        if (A_BF16)
            async_copy32((const unsigned short*)Ap + (size_t)(m0 + ar) * K + ah * 16,
                         &Alds[0][ldsOff]);
        async_copy32(Bt + (size_t)(n0 + ar) * K + ah * 16, &Blds[0][ldsOff]);
#else
        if (A_BF16) {
            const uint4* s = (const uint4*)((const unsigned short*)Ap +
                                            (size_t)(m0 + ar) * K + ah * 16);
            uint4* d = (uint4*)&Alds[0][ldsOff];
            d[0] = s[0]; d[1] = s[1];
        }
        {
            const uint4* s = (const uint4*)(Bt + (size_t)(n0 + ar) * K + ah * 16);
            uint4* d = (uint4*)&Blds[0][ldsOff];
            d[0] = s[0]; d[1] = s[1];
        }
#endif
        if (!A_BF16) {
            const float4* s = (const float4*)((const float*)Ap +
                                              (size_t)(m0 + ar) * K + ah * 16);
            unsigned short* d = &Alds[0][ldsOff];
#pragma unroll
            for (int i = 0; i < 4; ++i) {
                float4 v = s[i];
                d[i * 4 + 0] = f2bf(v.x); d[i * 4 + 1] = f2bf(v.y);
                d[i * 4 + 2] = f2bf(v.z); d[i * 4 + 3] = f2bf(v.w);
            }
        }
#if USE_ASYNC_LDS
        async_wait0();
#endif
        __syncthreads();
    }

    // ---------------- main pipeline ----------------
    for (int kt = 0; kt < kstages; ++kt) {
        const int cur = kt & 1, nxt = cur ^ 1;
        const int k1  = (kt + 1) * 32;
        const bool haveNext = (kt + 1 < kstages);

        float4 aReg0, aReg1, aReg2, aReg3;       // fp32 A staging
#if !USE_ASYNC_LDS
        uint4 aU0, aU1, bU0, bU1;                // bf16 staging fallback
#endif
        if (haveNext) {
#if USE_ASYNC_LDS
            if (A_BF16)
                async_copy32((const unsigned short*)Ap + (size_t)(m0 + ar) * K + k1 + ah * 16,
                             &Alds[nxt][ldsOff]);
            async_copy32(Bt + (size_t)(n0 + ar) * K + k1 + ah * 16, &Blds[nxt][ldsOff]);
#else
            if (A_BF16) {
                const uint4* s = (const uint4*)((const unsigned short*)Ap +
                                                (size_t)(m0 + ar) * K + k1 + ah * 16);
                aU0 = s[0]; aU1 = s[1];
            }
            {
                const uint4* s = (const uint4*)(Bt + (size_t)(n0 + ar) * K + k1 + ah * 16);
                bU0 = s[0]; bU1 = s[1];
            }
#endif
            if (!A_BF16) {
                const float4* s = (const float4*)((const float*)Ap +
                                                  (size_t)(m0 + ar) * K + k1 + ah * 16);
                aReg0 = s[0]; aReg1 = s[1]; aReg2 = s[2]; aReg3 = s[3];
            }
        }

        // ---- compute on current buffers ----
        const unsigned int* A32 = (const unsigned int*)Alds[cur];
        const unsigned int* B32 = (const unsigned int*)Blds[cur];
        union Frag { v16bf v; unsigned int u[8]; };
        Frag a[4], b[2];
#pragma unroll
        for (int tr = 0; tr < 4; ++tr) {
            int m = wr * 64 + tr * 16 + l16;
#pragma unroll
            for (int j = 0; j < 8; ++j) {
                int g = j >> 2, q = j & 3;
                // A 16-bit layout: K = g*16 + half*8 + 2q
                a[tr].u[j] = A32[m * 16 + g * 8 + half * 4 + q];
            }
        }
#pragma unroll
        for (int tc = 0; tc < 2; ++tc) {
            int n = wc * 32 + tc * 16 + l16;
#pragma unroll
            for (int j = 0; j < 8; ++j) {
                // B 16-bit layout: K = g*16 + half*8 + 2q
                b[tc].u[j] = B32[n * 16 + (j >> 2) * 8 + half * 4 + (j & 3)];
            }
        }
#pragma unroll
        for (int tr = 0; tr < 4; ++tr)
#pragma unroll
            for (int tc = 0; tc < 2; ++tc)
                acc[tr][tc] = __builtin_amdgcn_wmma_f32_16x16x32_bf16(
                    false, a[tr].v, false, b[tc].v, (short)0, acc[tr][tc],
                    false, false);

        // ---- finish staging next buffers ----
        if (haveNext) {
            if (!A_BF16) {
                unsigned short* d = &Alds[nxt][ldsOff];
                float4 r[4] = {aReg0, aReg1, aReg2, aReg3};
#pragma unroll
                for (int i = 0; i < 4; ++i) {
                    d[i * 4 + 0] = f2bf(r[i].x); d[i * 4 + 1] = f2bf(r[i].y);
                    d[i * 4 + 2] = f2bf(r[i].z); d[i * 4 + 3] = f2bf(r[i].w);
                }
            }
#if !USE_ASYNC_LDS
            if (A_BF16) {
                uint4* d = (uint4*)&Alds[nxt][ldsOff];
                d[0] = aU0; d[1] = aU1;
            }
            {
                uint4* d = (uint4*)&Blds[nxt][ldsOff];
                d[0] = bU0; d[1] = bU1;
            }
#else
            async_wait0();
#endif
        }
        __syncthreads();
    }

    // ---------------- epilogue ----------------
    const float PI = 3.14159265358979323846f;
#pragma unroll
    for (int tr = 0; tr < 4; ++tr) {
#pragma unroll
        for (int tc = 0; tc < 2; ++tc) {
#pragma unroll
            for (int v = 0; v < 8; ++v) {
                int m = m0 + wr * 64 + tr * 16 + v + 8 * half;
                int n = n0 + wc * 32 + tc * 16 + l16;
                float val = acc[tr][tc][v];
                if (EPI == EPI_ANG) {
                    float a = tanhf(val + bias[n]) * PI;
                    float nr = rintf(a * (3.0f / PI)) * (PI / 3.0f);
                    a = (fabsf(a - nr) < 0.15f) ? nr : a;
                    ((float*)Out)[(size_t)m * N + n] = a;
                } else if (EPI == EPI_AMP) {
                    float s = 1.0f / (1.0f + __expf(-(val + bias[n])));
                    ((float*)Out)[(size_t)m * N + n] = s;
                } else if (EPI == EPI_GELU) {
                    float x = val + bias[n];
                    float t = tanhf(0.7978845608f * (x + 0.044715f * x * x * x));
                    ((unsigned short*)Out)[(size_t)m * N + n] = f2bf(0.5f * x * (1.0f + t));
                } else if (EPI == EPI_MOE) {
                    float* O = (float*)Out;
                    O[(size_t)m * N + n] += we[(size_t)m * N_EXP] * (val + bias[n]);
                } else { // EPI_HEAD
                    ((float*)Out)[(size_t)m * N + n] = val + bias[n];
                }
            }
        }
    }
}

// ---------------------------------------------------------------------------
// Launch
// ---------------------------------------------------------------------------
extern "C" void kernel_launch(void* const* d_in, const int* in_sizes, int n_in,
                              void* d_out, int out_size, void* d_ws, size_t ws_size,
                              hipStream_t stream) {
    const int*   byte_seq = (const int*)  d_in[0];
    const float* emb      = (const float*)d_in[1];
    const float* phi_w    = (const float*)d_in[2];
    const float* phi_b    = (const float*)d_in[3];
    const float* amp_w    = (const float*)d_in[4];
    const float* amp_b    = (const float*)d_in[5];
    const float* gate_w   = (const float*)d_in[6];
    const float* gate_b   = (const float*)d_in[7];
    const float* w1       = (const float*)d_in[8];
    const float* b1       = (const float*)d_in[9];
    const float* w2       = (const float*)d_in[10];
    const float* b2       = (const float*)d_in[11];
    const float* head_w   = (const float*)d_in[12];
    const float* head_b   = (const float*)d_in[13];

    char* ws = (char*)d_ws;
    size_t off = 0;
    auto alloc = [&](size_t bytes) -> void* {
        void* p = ws + off;
        off += (bytes + 255) & ~(size_t)255;
        return p;
    };
    float*          ANG  = (float*)         alloc((size_t)NTOK * D_MODEL * 4);
    float*          AMP  = (float*)         alloc((size_t)NTOK * D_MODEL * 4);
    unsigned short* XE   = (unsigned short*)alloc((size_t)NTOK * D_MODEL * 2);
    unsigned short* X1   = (unsigned short*)alloc((size_t)NTOK * D_MODEL * 2);
    unsigned short* H    = (unsigned short*)alloc((size_t)NTOK * D_FF * 2);
    float*          OUT  = (float*)         alloc((size_t)NTOK * D_MODEL * 4);
    float*          WE   = (float*)         alloc((size_t)NTOK * N_EXP * 4);
    unsigned short* PHIT = (unsigned short*)alloc((size_t)D_MODEL * D_MODEL * 2);
    unsigned short* AMPT = (unsigned short*)alloc((size_t)D_MODEL * D_MODEL * 2);
    unsigned short* W1T  = (unsigned short*)alloc((size_t)N_EXP * D_FF * D_MODEL * 2);
    unsigned short* W2T  = (unsigned short*)alloc((size_t)N_EXP * D_MODEL * D_FF * 2);
    unsigned short* HDT  = (unsigned short*)alloc((size_t)VOCAB * D_MODEL * 2);

    float* logits = (float*)d_out;                         // [NTOK, VOCAB]
    float* nstate = logits + (size_t)NTOK * VOCAB;         // [BATCH, D_MODEL]

    // 0) weight pre-pass: transpose + convert to bf16 (Bt[n][k])
    transpose_to_bf16<<<(D_MODEL * D_MODEL) / 256, 256, 0, stream>>>(phi_w, PHIT, D_MODEL, D_MODEL);
    transpose_to_bf16<<<(D_MODEL * D_MODEL) / 256, 256, 0, stream>>>(amp_w, AMPT, D_MODEL, D_MODEL);
    transpose_to_bf16<<<(N_EXP * D_FF * D_MODEL) / 256, 256, 0, stream>>>(w1, W1T, D_MODEL, D_FF);
    transpose_to_bf16<<<(N_EXP * D_MODEL * D_FF) / 256, 256, 0, stream>>>(w2, W2T, D_FF, D_MODEL);
    transpose_to_bf16<<<(VOCAB * D_MODEL) / 256, 256, 0, stream>>>(head_w, HDT, D_MODEL, VOCAB);

    // 1) embedding -> bf16 X
    embed_kernel<<<(NTOK * D_MODEL) / 256, 256, 0, stream>>>(byte_seq, emb, XE);

    // 2) angle / amplitude GEMMs (parallel over all tokens; recurrence-free)
    gemm_wmma<true, EPI_ANG><<<dim3(D_MODEL / 128, NTOK / 128), 256, 0, stream>>>(
        XE, PHIT, phi_b, nullptr, ANG, NTOK, D_MODEL, D_MODEL);
    gemm_wmma<true, EPI_AMP><<<dim3(D_MODEL / 128, NTOK / 128), 256, 0, stream>>>(
        XE, AMPT, amp_b, nullptr, AMP, NTOK, D_MODEL, D_MODEL);

    // 3) elementwise toroidal scan (the only sequential dependency)
    scan_kernel<<<(BATCH * D_MODEL) / 256, 256, 0, stream>>>(ANG, AMP, X1, nstate);

    // 4) top-2 routing weights
    gate_kernel<<<NTOK / 256, 256, 0, stream>>>(X1, gate_w, gate_b, WE);

    // 5) MoE: dense per-expert FFN, WMMA both GEMMs
    hipMemsetAsync(OUT, 0, (size_t)NTOK * D_MODEL * 4, stream);
    for (int e = 0; e < N_EXP; ++e) {
        gemm_wmma<true, EPI_GELU><<<dim3(D_FF / 128, NTOK / 128), 256, 0, stream>>>(
            X1, W1T + (size_t)e * D_FF * D_MODEL, b1 + (size_t)e * D_FF,
            nullptr, H, NTOK, D_FF, D_MODEL);
        gemm_wmma<true, EPI_MOE><<<dim3(D_MODEL / 128, NTOK / 128), 256, 0, stream>>>(
            H, W2T + (size_t)e * D_MODEL * D_FF, b2 + (size_t)e * D_MODEL,
            WE + e, OUT, NTOK, D_MODEL, D_FF);
    }

    // 6) head projection (A is fp32 OUT, converted on the LDS-staging path)
    gemm_wmma<false, EPI_HEAD><<<dim3(VOCAB / 128, NTOK / 128), 256, 0, stream>>>(
        OUT, HDT, head_b, nullptr, logits, NTOK, VOCAB, D_MODEL);
}
